// LocalPatternExtractor_82222853915140
// MI455X (gfx1250) — compile-verified, matrix operand order checked by reading, not run on hardware
//
#include <hip/hip_runtime.h>

// ---------------------------------------------------------------------------
// Binary-weight depthwise+pointwise conv -> BN -> quantized LIF (4 steps)
// MI455X (gfx1250, wave32). Pointwise GEMM on V_WMMA_F32_16X16X32_F16 with
// sign-only A (exact in f16) and hi/lo-split f16 B (fp32-accurate product).
// BN stats via 12x12 second-moment factorization (no per-channel reductions).
// Memory-bound: ~86 MB traffic -> ~3.7us floor at 23.3 TB/s.
// ---------------------------------------------------------------------------

typedef __attribute__((ext_vector_type(16))) _Float16 v16h;
typedef __attribute__((ext_vector_type(8)))  _Float16 v8h;
typedef __attribute__((ext_vector_type(8)))  float    v8f;

#define B_SZ      16
#define C_IN      12
#define L_SZ      5000
#define C_OUT     256
#define P_TOT     (B_SZ * L_SZ)         // 80000 positions
#define POS_BLK   128                   // positions per block tile (8 waves x 16)
#define NBLK_MAIN 625                   // 625*128 = 80000 exactly
#define NBLK_STAT 20
#define NSTAT_VAL 90                    // 12 sums + 78 upper-tri second moments

// workspace byte offsets
#define OFF_A     0                     // 256 rows * 32 halfs (sign A frags) = 16384 B
#define OFF_WDB   16384                 // 36 floats: binarized depthwise weights
#define OFF_SCL   16640                 // 256 floats: pointwise |w| scale
#define OFF_ACOEF 17664                 // 256 floats: 128*gamma*rsqrt(var+eps)*scale
#define OFF_BCOEF 18688                 // 256 floats: 128*(beta - gamma*rsqrt*mu)
#define OFF_STATS 19712                 // 20 blocks * 90 floats
#define OFF_REG   26912                 // 625 floats reg-loss partials

__device__ __forceinline__ float sgnf(float w) {
    return (w > 0.f) ? 1.f : ((w < 0.f) ? -1.f : 0.f);
}

__device__ __forceinline__ void depthwise12(const float* __restrict__ x,
                                            const float* __restrict__ wdb,
                                            int p, float* yd) {
    int b = p / L_SZ;
    int l = p - b * L_SZ;
    #pragma unroll
    for (int c = 0; c < C_IN; ++c) {
        const float* xb = x + (size_t)(b * C_IN + c) * L_SZ;
        float xm = (l > 0)        ? xb[l - 1] : 0.f;
        float x0 = xb[l];
        float xp = (l < L_SZ - 1) ? xb[l + 1] : 0.f;
        yd[c] = wdb[c * 3] * xm + wdb[c * 3 + 1] * x0 + wdb[c * 3 + 2] * xp;
    }
}

// --------------------------------------------------------------------------
// Stage depthwise output for 128 positions into LDS in WMMA-B layout:
// row n (64 B): halfs [0..11] = hi(yd[c]), [12..15]=0, [16..27]=lo, [28..31]=0
// --------------------------------------------------------------------------
__device__ __forceinline__ void stage_yd(const float* __restrict__ x,
                                         const float* __restrict__ wdb,
                                         _Float16* ydL, int p0, int t) {
    #pragma unroll
    for (int e = t; e < POS_BLK * 4; e += 256) {      // zero pad dwords 6,7,14,15
        int n = e >> 2, q = e & 3;
        int dwi = (q < 2) ? (6 + q) : (12 + q);
        ((unsigned int*)ydL)[n * 16 + dwi] = 0u;
    }
    #pragma unroll
    for (int e = t; e < C_IN * POS_BLK; e += 256) {
        int c = e >> 7, n = e & (POS_BLK - 1);
        int p = p0 + n;
        int b = p / L_SZ;
        int l = p - b * L_SZ;
        const float* xb = x + (size_t)(b * C_IN + c) * L_SZ;
        float xm = (l > 0)        ? xb[l - 1] : 0.f;
        float x0 = xb[l];
        float xp = (l < L_SZ - 1) ? xb[l + 1] : 0.f;
        float v  = wdb[c * 3] * xm + wdb[c * 3 + 1] * x0 + wdb[c * 3 + 2] * xp;
        _Float16 hi = (_Float16)v;
        _Float16 lo = (_Float16)(v - (float)hi);
        ydL[n * 32 + c]      = hi;
        ydL[n * 32 + 16 + c] = lo;
    }
}

#define SHUF16(a, b) __builtin_shufflevector(a, b, 0,1,2,3,4,5,6,7,8,9,10,11,12,13,14,15)

// --------------------------------------------------------------------------
// prep: binarize weights, build sign-only A fragments, pointwise scales
// --------------------------------------------------------------------------
__global__ void prep_kernel(const float* __restrict__ dw,
                            const float* __restrict__ pw, void* wsv) {
    char* ws = (char*)wsv;
    _Float16* wsA = (_Float16*)(ws + OFF_A);
    float* wdb = (float*)(ws + OFF_WDB);
    float* scl = (float*)(ws + OFF_SCL);
    int t = threadIdx.x;
    if (t < C_IN) {
        float w0 = dw[t * 3], w1 = dw[t * 3 + 1], w2 = dw[t * 3 + 2];
        float s = (fabsf(w0) + fabsf(w1) + fabsf(w2)) * (1.f / 3.f);
        wdb[t * 3 + 0] = sgnf(w0) * s;
        wdb[t * 3 + 1] = sgnf(w1) * s;
        wdb[t * 3 + 2] = sgnf(w2) * s;
    }
    {
        float s = 0.f;
        #pragma unroll
        for (int c = 0; c < C_IN; ++c) s += fabsf(pw[t * C_IN + c]);
        scl[t] = s * (1.f / (float)C_IN);
    }
    // A fragment: row r, half h: halfs[0..7]=sign(c=h*8+s), halfs[8..15]=same (lo K)
    #pragma unroll
    for (int e = t; e < C_OUT * 32; e += 256) {
        int r = e >> 5, idx = e & 31;
        int h = (idx >> 4), sl = idx & 15;
        int c = h * 8 + (sl & 7);
        float w = (c < C_IN) ? pw[r * C_IN + c] : 0.f;
        wsA[e] = (_Float16)sgnf(w);
    }
}

// --------------------------------------------------------------------------
// pass 1: S[c] = sum yd_c, M[c,c'] = sum yd_c*yd_c' (upper tri, 90 values).
// Pure-register accumulation; deterministic wave->block->global reduction.
// --------------------------------------------------------------------------
__global__ void stats_kernel(const float* __restrict__ x, void* wsv) {
    char* ws = (char*)wsv;
    const float* wdb = (const float*)(ws + OFF_WDB);
    float* statP = (float*)(ws + OFF_STATS);

    __shared__ float sWave[8 * NSTAT_VAL];

    int t = threadIdx.x;
    int lane = t & 31, wv = t >> 5;
    int gid = blockIdx.x * 256 + t;
    const int stride = NBLK_STAT * 256;

    float S[C_IN];
    float M[78];
    #pragma unroll
    for (int c = 0; c < C_IN; ++c) S[c] = 0.f;
    #pragma unroll
    for (int v = 0; v < 78; ++v) M[v] = 0.f;

    for (int p = gid; p < P_TOT; p += stride) {
        float yd[C_IN];
        depthwise12(x, wdb, p, yd);
        int idx = 0;
        #pragma unroll
        for (int c = 0; c < C_IN; ++c) {
            S[c] += yd[c];
            #pragma unroll
            for (int cc = c; cc < C_IN; ++cc) {
                M[idx] = fmaf(yd[c], yd[cc], M[idx]);
                ++idx;
            }
        }
    }

    // wave reduce each of the 90 values, lane0 -> per-wave LDS slot
    #pragma unroll
    for (int v = 0; v < NSTAT_VAL; ++v) {
        float s = (v < C_IN) ? S[v] : M[v - C_IN];
        #pragma unroll
        for (int off = 1; off < 32; off <<= 1) s += __shfl_xor(s, off, 32);
        if (lane == 0) sWave[wv * NSTAT_VAL + v] = s;
    }
    __syncthreads();
    if (t < NSTAT_VAL) {
        float s = 0.f;
        #pragma unroll
        for (int w = 0; w < 8; ++w) s += sWave[w * NSTAT_VAL + t];
        statP[blockIdx.x * NSTAT_VAL + t] = s;
    }
}

// --------------------------------------------------------------------------
// reduce: total moments -> per-channel mean/var via quadratic form -> BN
// coefficients prescaled by 128 (quantization-step units for the LIF).
// --------------------------------------------------------------------------
__global__ void reduce_kernel(const float* __restrict__ pw,
                              const float* __restrict__ gamma,
                              const float* __restrict__ beta, void* wsv) {
    char* ws = (char*)wsv;
    const float* statP = (const float*)(ws + OFF_STATS);
    const float* scl = (const float*)(ws + OFF_SCL);
    float* aC = (float*)(ws + OFF_ACOEF);
    float* bC = (float*)(ws + OFF_BCOEF);

    __shared__ float tot[NSTAT_VAL];
    int t = threadIdx.x;
    if (t < NSTAT_VAL) {
        float s = 0.f;
        for (int bk = 0; bk < NBLK_STAT; ++bk) s += statP[bk * NSTAT_VAL + t];
        tot[t] = s;
    }
    __syncthreads();

    int o = t;
    float sg[C_IN];
    #pragma unroll
    for (int c = 0; c < C_IN; ++c) sg[c] = sgnf(pw[o * C_IN + c]);
    float sc = scl[o];

    float s1 = 0.f;
    #pragma unroll
    for (int c = 0; c < C_IN; ++c) s1 = fmaf(sg[c], tot[c], s1);
    s1 *= sc;

    float s2 = 0.f;
    int idx = C_IN;
    #pragma unroll
    for (int c = 0; c < C_IN; ++c) {
        #pragma unroll
        for (int cc = c; cc < C_IN; ++cc) {
            float w = sg[c] * sg[cc] * tot[idx];
            s2 += (cc == c) ? w : 2.f * w;
            ++idx;
        }
    }
    s2 *= sc * sc;

    float inv = 1.f / (float)P_TOT;
    float mu  = s1 * inv;
    float var = s2 * inv - mu * mu;
    float rs  = rsqrtf(var + 1e-5f);
    float a   = gamma[o] * rs;
    aC[o] = 128.f * a * sc;                  // applied to unscaled WMMA sum
    bC[o] = 128.f * (beta[o] - a * mu);      // in quantization-step units
}

// --------------------------------------------------------------------------
// pass 2: y via WMMA, BN, quantized LIF x4 (step units), store + reg partials
// --------------------------------------------------------------------------
__global__ void main_kernel(const float* __restrict__ x, void* wsv,
                            float* __restrict__ out) {
    char* ws = (char*)wsv;
    const _Float16* wsA = (const _Float16*)(ws + OFF_A);
    const float* wdb = (const float*)(ws + OFF_WDB);
    const float* aC = (const float*)(ws + OFF_ACOEF);
    const float* bC = (const float*)(ws + OFF_BCOEF);
    float* regP = (float*)(ws + OFF_REG);

    __shared__ __align__(16) _Float16 ydL[POS_BLK * 32];
    __shared__ float sA[C_OUT], sB[C_OUT];
    __shared__ float sWaveReg[8];

    int t = threadIdx.x;
    sA[t] = aC[t]; sB[t] = bC[t];

    int p0 = blockIdx.x * POS_BLK;
    stage_yd(x, wdb, ydL, p0, t);
    __syncthreads();

    int lane = t & 31, wv = t >> 5;
    int half = lane >> 4, m = lane & 15;
    int nloc = (wv << 4) + m;
    int p = p0 + nloc;
    int b = p / L_SZ;
    int l = p - b * L_SZ;

    v8h blo = *(const v8h*)(ydL + nloc * 32 + half * 8);
    v8h bhi = *(const v8h*)(ydL + nloc * 32 + 16 + half * 8);
    v16h bfrag = SHUF16(blo, bhi);

    float regsum = 0.f;
    for (int mt = 0; mt < 16; ++mt) {
        const v8h* ap = (const v8h*)(wsA + ((((mt * 16 + m) * 2) + half) << 4));
        v16h afrag = SHUF16(ap[0], ap[1]);
        v8f acc = {0.f, 0.f, 0.f, 0.f, 0.f, 0.f, 0.f, 0.f};
        acc = __builtin_amdgcn_wmma_f32_16x16x32_f16(
            false, afrag, false, bfrag, (short)0, acc, false, false);
        #pragma unroll
        for (int j = 0; j < 8; ++j) {
            int o = mt * 16 + half * 8 + j;
            float yb = fmaf(acc[j], sA[o], sB[o]);   // in units of step=1/128
            // quantized LIF, decay 0.5, soft reset; threshold = 128 units
            float mem = 0.f, cnt = 0.f;
            #pragma unroll
            for (int ts = 0; ts < 4; ++ts) {
                mem = fmaf(0.5f, mem, yb);
                mem = rintf(mem);
                mem = fmaxf(-128.f, fminf(mem, 127.f));
                float spk = (mem >= 128.f) ? 1.f : 0.f;
                mem = fmaf(-128.f, spk, mem);
                cnt += spk;
            }
            float ov = cnt * 0.25f;
            out[(size_t)(b * C_OUT + o) * L_SZ + l] = ov;
            regsum += ov;
        }
    }
    #pragma unroll
    for (int off = 1; off < 32; off <<= 1) regsum += __shfl_xor(regsum, off, 32);
    if (lane == 0) sWaveReg[wv] = regsum;
    __syncthreads();
    if (t == 0) {
        float s = 0.f;
        #pragma unroll
        for (int w = 0; w < 8; ++w) s += sWaveReg[w];
        regP[blockIdx.x] = s;
    }
}

// --------------------------------------------------------------------------
// reg_loss = SPIKE_REG * mean(out), deterministic serial reduction
// --------------------------------------------------------------------------
__global__ void final_kernel(void* wsv, float* __restrict__ out, int outN) {
    if (threadIdx.x == 0 && blockIdx.x == 0) {
        const float* regP = (const float*)((char*)wsv + OFF_REG);
        float s = 0.f;
        for (int i = 0; i < NBLK_MAIN; ++i) s += regP[i];
        out[outN] = 0.01f * s / (float)outN;
    }
}

extern "C" void kernel_launch(void* const* d_in, const int* in_sizes, int n_in,
                              void* d_out, int out_size, void* d_ws, size_t ws_size,
                              hipStream_t stream) {
    const float* x     = (const float*)d_in[0];
    const float* dw    = (const float*)d_in[1];
    const float* pw    = (const float*)d_in[2];
    const float* gamma = (const float*)d_in[3];
    const float* beta  = (const float*)d_in[4];
    float* out = (float*)d_out;

    prep_kernel  <<<1,         256, 0, stream>>>(dw, pw, d_ws);
    stats_kernel <<<NBLK_STAT, 256, 0, stream>>>(x, d_ws);
    reduce_kernel<<<1,         256, 0, stream>>>(pw, gamma, beta, d_ws);
    main_kernel  <<<NBLK_MAIN, 256, 0, stream>>>(x, d_ws, out);
    final_kernel <<<1,          64, 0, stream>>>(d_ws, out, out_size - 1);
}
